// minGRU_12678743458348
// MI455X (gfx1250) — matrix operand order repaired and checked by reading, compiled, not run on hardware
//
#include <hip/hip_runtime.h>
#include <hip/hip_bf16.h>
#include <math.h>

// ---------------------------------------------------------------------------
// Types for CDNA5 WMMA (wave32): v_wmma_f32_16x16x32_bf16
// ---------------------------------------------------------------------------
typedef __attribute__((ext_vector_type(16))) __bf16 v16bf;
typedef __attribute__((ext_vector_type(8)))  float  v8f;
typedef __attribute__((ext_vector_type(4)))  float  f4_t;

union FragAB {
    v16bf v;
    f4_t  f[2];   // 2 x 16 bytes = 16 bf16
};

// Problem constants: x[B=4, S=4096, D=1024], W[2D=2048, D=1024]
#define S_LEN   4096
#define D_DIM   1024
#define M_TOT   16384          // B*S
#define K_TOT   1024
#define N_HID   1024
#define N_CHAN  4096           // B*D channels for the scan
#define CHUNKS  64
#define CH_LEN  64             // S_LEN / CHUNKS

// ---------------------------------------------------------------------------
// Math helpers (match jax.nn.softplus / cumlogsumexp semantics, fp32)
// ---------------------------------------------------------------------------
__device__ __forceinline__ float softplusf(float x) {
    // log(1 + exp(x)), numerically stable
    return fmaxf(x, 0.0f) + log1pf(expf(-fabsf(x)));
}
__device__ __forceinline__ float log_g(float x) {
    // log(x + 0.5) for x >= 0, log(sigmoid(x)) for x < 0
    return (x >= 0.0f) ? logf(x + 0.5f) : -softplusf(-x);
}
__device__ __forceinline__ float logaddexpf_(float a, float b) {
    float m = fmaxf(a, b);
    if (m == -INFINITY) return -INFINITY;
    return m + log1pf(expf(fminf(a, b) - m));
}

// ---------------------------------------------------------------------------
// Kernel 1: split fp32 -> (bf16 hi, bf16 lo) for Markidis 3-pass GEMM
// ---------------------------------------------------------------------------
__global__ void cvt_split_kernel(const float* __restrict__ src,
                                 __hip_bfloat16* __restrict__ hi,
                                 __hip_bfloat16* __restrict__ lo,
                                 long n) {
    long i = (long)blockIdx.x * blockDim.x + threadIdx.x;
    if (i >= n) return;
    float v = src[i];
    __hip_bfloat16 h = __float2bfloat16(v);
    hi[i] = h;
    lo[i] = __float2bfloat16(v - __bfloat162float(h));
}

// ---------------------------------------------------------------------------
// Fragment loaders per CDNA5 ISA 7.12.2 (wave32, 16-bit operands)
// A 16x32: lane L holds row M=L%16; K chunks {0..7,16..23} (L<16) /
//          {8..15,24..31} (L>=16)  -> two 16B loads at kb and kb+16
// B 32x16: lane L holds col N=L%16; K chunks {0..7,16..23} (L<16) /
//          {8..15,24..31} (L>=16)  -> two 16B loads at kb and kb+16
// Both x (A, rows of x) and W (B, rows of W, since C = x * W^T) are
// row-major with K-stride 1024, so the address math is shared.
// ---------------------------------------------------------------------------
__device__ __forceinline__ void load_fragA(FragAB& fr, const __bf16* base,
                                           int row, int k0, int lane) {
    int kb = k0 + ((lane & 16) ? 8 : 0);
    const __bf16* p = base + (size_t)row * K_TOT + kb;
    fr.f[0] = *(const f4_t*)(p);
    fr.f[1] = *(const f4_t*)(p + 16);
}
__device__ __forceinline__ void load_fragB(FragAB& fr, const __bf16* base,
                                           int row, int k0, int lane) {
    int kb = k0 + ((lane & 16) ? 8 : 0);
    const __bf16* p = base + (size_t)row * K_TOT + kb;
    fr.f[0] = *(const f4_t*)(p);
    fr.f[1] = *(const f4_t*)(p + 16);
}

// ---------------------------------------------------------------------------
// Kernel 2: fused GEMM (hg = x * W^T) + minGRU log-space epilogue.
// Each block: 8 waves, block tile M=64 x N=128 over the *hidden* columns,
// and simultaneously the matching gate columns (n+1024) so the epilogue can
// compute log_coeffs (LA) and log_values (LV) element-wise.
// 3-pass bf16 split accumulated in fp32: AhBh + AhBl + AlBh.
// ---------------------------------------------------------------------------
__global__ void __launch_bounds__(256)
gemm_mingru_kernel(const __hip_bfloat16* __restrict__ xh_,
                   const __hip_bfloat16* __restrict__ xl_,
                   const __hip_bfloat16* __restrict__ wh_,
                   const __hip_bfloat16* __restrict__ wl_,
                   float* __restrict__ LA,
                   float* __restrict__ LV) {
    const __bf16* Xh = (const __bf16*)xh_;
    const __bf16* Xl = (const __bf16*)xl_;
    const __bf16* Wh = (const __bf16*)wh_;
    const __bf16* Wl = (const __bf16*)wl_;

    const int lane  = threadIdx.x & 31;
    const int wave  = threadIdx.x >> 5;          // 0..7
    const int waveM = wave >> 2;                 // 0..1
    const int waveN = wave & 3;                  // 0..3
    const int m0 = blockIdx.x * 64 + waveM * 32; // wave M base
    const int n0 = blockIdx.y * 128 + waveN * 32;// wave N base (hidden cols)
    const int lm = lane & 15;

    v8f accH[2][2] = {};   // hidden accumulators
    v8f accG[2][2] = {};   // gate accumulators

    for (int k0 = 0; k0 < K_TOT; k0 += 32) {
        FragAB ah[2], al[2];
#pragma unroll
        for (int mt = 0; mt < 2; ++mt) {
            int row = m0 + mt * 16 + lm;
            load_fragA(ah[mt], Xh, row, k0, lane);
            load_fragA(al[mt], Xl, row, k0, lane);
        }
        FragAB bhH[2], blH[2], bhG[2], blG[2];
#pragma unroll
        for (int nt = 0; nt < 2; ++nt) {
            int nrow = n0 + nt * 16 + lm;
            load_fragB(bhH[nt], Wh, nrow,          k0, lane);
            load_fragB(blH[nt], Wl, nrow,          k0, lane);
            load_fragB(bhG[nt], Wh, nrow + N_HID,  k0, lane);
            load_fragB(blG[nt], Wl, nrow + N_HID,  k0, lane);
        }
#pragma unroll
        for (int mt = 0; mt < 2; ++mt) {
#pragma unroll
            for (int nt = 0; nt < 2; ++nt) {
                // hidden: hi*hi + hi*lo + lo*hi
                accH[mt][nt] = __builtin_amdgcn_wmma_f32_16x16x32_bf16(
                    false, ah[mt].v, false, bhH[nt].v, (short)0, accH[mt][nt], false, false);
                accH[mt][nt] = __builtin_amdgcn_wmma_f32_16x16x32_bf16(
                    false, ah[mt].v, false, blH[nt].v, (short)0, accH[mt][nt], false, false);
                accH[mt][nt] = __builtin_amdgcn_wmma_f32_16x16x32_bf16(
                    false, al[mt].v, false, bhH[nt].v, (short)0, accH[mt][nt], false, false);
                // gate
                accG[mt][nt] = __builtin_amdgcn_wmma_f32_16x16x32_bf16(
                    false, ah[mt].v, false, bhG[nt].v, (short)0, accG[mt][nt], false, false);
                accG[mt][nt] = __builtin_amdgcn_wmma_f32_16x16x32_bf16(
                    false, ah[mt].v, false, blG[nt].v, (short)0, accG[mt][nt], false, false);
                accG[mt][nt] = __builtin_amdgcn_wmma_f32_16x16x32_bf16(
                    false, al[mt].v, false, bhG[nt].v, (short)0, accG[mt][nt], false, false);
            }
        }
    }

    // Epilogue: C/D layout -> element (M = r + (lane>=16 ? 8 : 0), N = lane%16)
    const int mhi = (lane & 16) ? 8 : 0;
#pragma unroll
    for (int mt = 0; mt < 2; ++mt) {
#pragma unroll
        for (int nt = 0; nt < 2; ++nt) {
            int nn = n0 + nt * 16 + lm;
#pragma unroll
            for (int r = 0; r < 8; ++r) {
                int mm = m0 + mt * 16 + r + mhi;
                float h = accH[mt][nt][r];
                float g = accG[mt][nt][r];
                float la = -softplusf(g);                    // log(1 - sigmoid(g))
                float lv = -softplusf(-g) + log_g(h);        // log(sigmoid(g)*g(h))
                size_t idx = (size_t)mm * N_HID + nn;
                LA[idx] = la;
                LV[idx] = lv;
            }
        }
    }
}

// ---------------------------------------------------------------------------
// Scan monoid: (a1,v1) o (a2,v2) = (a1+a2, logaddexp(v1+a2, v2))
// Phase A: per-(channel, chunk) summaries (coalesced over channels)
// ---------------------------------------------------------------------------
__global__ void scan_chunks_kernel(const float* __restrict__ LA,
                                   const float* __restrict__ LV,
                                   float* __restrict__ SA,
                                   float* __restrict__ SV) {
    int j = blockIdx.x >> 4;                        // chunk 0..63
    int c = ((blockIdx.x & 15) << 8) + threadIdx.x; // channel 0..4095
    int b = c >> 10, d = c & (D_DIM - 1);
    size_t rowBase = ((size_t)b * S_LEN + (size_t)j * CH_LEN) * D_DIM + d;
    float A = 0.0f, V = -INFINITY;
    for (int s = 0; s < CH_LEN; ++s) {
        float la = LA[rowBase + (size_t)s * D_DIM];
        float lv = LV[rowBase + (size_t)s * D_DIM];
        A += la;
        V = logaddexpf_(V + la, lv);
    }
    SA[(size_t)j * N_CHAN + c] = A;
    SV[(size_t)j * N_CHAN + c] = V;
}

// Phase B: exclusive scan of chunk summaries along chunks (per channel)
__global__ void scan_prefix_kernel(const float* __restrict__ SA,
                                   const float* __restrict__ SV,
                                   float* __restrict__ PV) {
    int c = blockIdx.x * 256 + threadIdx.x;  // channel 0..4095
    float Vp = -INFINITY;
    for (int j = 0; j < CHUNKS; ++j) {
        PV[(size_t)j * N_CHAN + c] = Vp;     // exclusive prefix (V only needed)
        float Aj = SA[(size_t)j * N_CHAN + c];
        float Vj = SV[(size_t)j * N_CHAN + c];
        Vp = logaddexpf_(Vp + Aj, Vj);
    }
}

// Phase C: replay chunk with its prefix, write out = exp(log_h)
__global__ void scan_apply_kernel(const float* __restrict__ LA,
                                  const float* __restrict__ LV,
                                  const float* __restrict__ PV,
                                  float* __restrict__ out) {
    int j = blockIdx.x >> 4;
    int c = ((blockIdx.x & 15) << 8) + threadIdx.x;
    int b = c >> 10, d = c & (D_DIM - 1);
    size_t rowBase = ((size_t)b * S_LEN + (size_t)j * CH_LEN) * D_DIM + d;
    float V = PV[(size_t)j * N_CHAN + c];
    for (int s = 0; s < CH_LEN; ++s) {
        size_t idx = rowBase + (size_t)s * D_DIM;
        V = logaddexpf_(V + LA[idx], LV[idx]);
        out[idx] = expf(V);
    }
}

// ---------------------------------------------------------------------------
// Launch
// Workspace layout (bytes):
//   LA  @ 0          64 MB   log_coeffs plane [16384 x 1024]
//   LV  @ 64 MB      64 MB   log_values plane
//   xh  @ 128 MB     32 MB   bf16 hi(x)
//   xl  @ 160 MB     32 MB   bf16 lo(x)
//   wh  @ 192 MB      4 MB   bf16 hi(W)
//   wl  @ 196 MB      4 MB   bf16 lo(W)
//   SA  @ 200 MB      1 MB   chunk sums of log_coeffs  [64 x 4096]
//   SV  @ 201 MB      1 MB   chunk scan values
//   PV  @ 202 MB      1 MB   exclusive chunk prefixes
// ---------------------------------------------------------------------------
extern "C" void kernel_launch(void* const* d_in, const int* in_sizes, int n_in,
                              void* d_out, int out_size, void* d_ws, size_t ws_size,
                              hipStream_t stream) {
    (void)in_sizes; (void)n_in; (void)out_size; (void)ws_size;
    const float* x = (const float*)d_in[0];   // [4, 4096, 1024]
    const float* W = (const float*)d_in[1];   // [2048, 1024]
    float* out = (float*)d_out;               // [4, 4096, 1024]

    char* ws = (char*)d_ws;
    const size_t MB = 1024ull * 1024ull;
    float*          LA = (float*)(ws + 0);
    float*          LV = (float*)(ws + 64 * MB);
    __hip_bfloat16* xh = (__hip_bfloat16*)(ws + 128 * MB);
    __hip_bfloat16* xl = (__hip_bfloat16*)(ws + 160 * MB);
    __hip_bfloat16* wh = (__hip_bfloat16*)(ws + 192 * MB);
    __hip_bfloat16* wl = (__hip_bfloat16*)(ws + 196 * MB);
    float*          SA = (float*)(ws + 200 * MB);
    float*          SV = (float*)(ws + 201 * MB);
    float*          PV = (float*)(ws + 202 * MB);

    const long nx = (long)M_TOT * K_TOT;        // 16,777,216
    const long nw = (long)2 * N_HID * K_TOT;    //  2,097,152

    cvt_split_kernel<<<(unsigned)((nx + 255) / 256), 256, 0, stream>>>(x, xh, xl, nx);
    cvt_split_kernel<<<(unsigned)((nw + 255) / 256), 256, 0, stream>>>(W, wh, wl, nw);

    dim3 ggrid(M_TOT / 64, N_HID / 128);        // 256 x 8 blocks
    gemm_mingru_kernel<<<ggrid, 256, 0, stream>>>(xh, xl, wh, wl, LA, LV);

    scan_chunks_kernel<<<CHUNKS * 16, 256, 0, stream>>>(LA, LV, SA, SV);
    scan_prefix_kernel<<<N_CHAN / 256, 256, 0, stream>>>(SA, SV, PV);
    scan_apply_kernel<<<CHUNKS * 16, 256, 0, stream>>>(LA, LV, PV, out);
}
